// PairWeightedAveraging_16793322127900
// MI455X (gfx1250) — matrix operand order repaired and checked by reading, not compiled, not run on hardware
//
#include <hip/hip_runtime.h>
#include <hip/hip_bf16.h>

#define S_DIM 512
#define N_DIM 768
#define CM_DIM 64
#define CZ_DIM 128
#define H_DIM 8
#define D_DIM 32
#define HD_DIM 256
#define EPSV 1e-5f

typedef __attribute__((ext_vector_type(16))) _Float16 v16h;
typedef __attribute__((ext_vector_type(8)))  _Float16 v8h;
typedef __attribute__((ext_vector_type(8)))  float    v8f;

// ---------------------------------------------------------------------------
// CDNA5 async global->LDS copy (ASYNCcnt-tracked), with inline-asm fallback.
// Builtin signature (from clang diagnostic): params are pointers to
// 'int __attribute__((vector_size(16)))' -- global (as1) src, LDS (as3) dst.
// ---------------------------------------------------------------------------
#if __has_builtin(__builtin_amdgcn_global_load_async_to_lds_b128)
#define HAVE_ASYNC_LDS 1
#else
#define HAVE_ASYNC_LDS 0
#endif

__device__ __forceinline__ void async_copy_b128(const void* gsrc, void* ldst)
{
#if HAVE_ASYNC_LDS
    typedef int v4i_gcc __attribute__((vector_size(16)));
    typedef __attribute__((address_space(1))) v4i_gcc* as1p;
    typedef __attribute__((address_space(3))) v4i_gcc* as3p;
    __builtin_amdgcn_global_load_async_to_lds_b128(
        (as1p)(unsigned long long)gsrc,
        (as3p)(unsigned)(unsigned long long)ldst,
        0, 0);
#else
    asm volatile("global_load_async_to_lds_b128 %0, %1, off"
                 :: "v"((unsigned)(unsigned long long)ldst), "v"(gsrc)
                 : "memory");
#endif
}

#if __has_builtin(__builtin_amdgcn_s_wait_asynccnt)
#define WAIT_ASYNC(n) __builtin_amdgcn_s_wait_asynccnt(n)
#else
#define WAIT_ASYNC(n) asm volatile("s_wait_asynccnt %0" :: "i"(n) : "memory")
#endif

// ---------------------------------------------------------------------------
// Kernel 1: LayerNorm(m) + v = mn@wm (f16), g = sigmoid(mn@wg) (f16)
// vT layout: [h][s*D + d][p=n]  (K-contiguous rows for the big GEMM A operand)
// g16 layout: [s][q=n][h*D + d]
// One wave per (s,n) row; CM=64 -> 2 elements per lane; broadcast via shuffles.
// ---------------------------------------------------------------------------
__global__ __launch_bounds__(256) void ln_project_m_kernel(
    const float* __restrict__ m, const float* __restrict__ nw,
    const float* __restrict__ nb, const float* __restrict__ wm,
    const float* __restrict__ wg, _Float16* __restrict__ vT,
    _Float16* __restrict__ g16)
{
    const int lane = threadIdx.x & 31;
    const int wave = threadIdx.x >> 5;
    const int s    = blockIdx.x;

    const float w0 = nw[lane], w1 = nw[lane + 32];
    const float b0 = nb[lane], b1 = nb[lane + 32];

    for (int n = wave; n < N_DIM; n += 8) {
        const float* row = m + ((size_t)s * N_DIM + n) * CM_DIM;
        float x0 = row[lane], x1 = row[lane + 32];

        float sum = x0 + x1;
        #pragma unroll
        for (int off = 16; off >= 1; off >>= 1) sum += __shfl_xor(sum, off, 32);
        float mu = sum * (1.0f / CM_DIM);

        float d0 = x0 - mu, d1 = x1 - mu;
        float vs = d0 * d0 + d1 * d1;
        #pragma unroll
        for (int off = 16; off >= 1; off >>= 1) vs += __shfl_xor(vs, off, 32);
        float rstd = rsqrtf(vs * (1.0f / CM_DIM) + EPSV);

        float xn0 = d0 * rstd * w0 + b0;
        float xn1 = d1 * rstd * w1 + b1;

        float av[8], ag[8];
        #pragma unroll
        for (int j = 0; j < 8; ++j) { av[j] = 0.f; ag[j] = 0.f; }

        for (int c = 0; c < CM_DIM; ++c) {
            float mnc = (c < 32) ? __shfl(xn0, c, 32) : __shfl(xn1, c - 32, 32);
            const float* wmr = wm + (size_t)c * HD_DIM;
            const float* wgr = wg + (size_t)c * HD_DIM;
            #pragma unroll
            for (int j = 0; j < 8; ++j) {
                int k = lane + 32 * j;
                av[j] = fmaf(mnc, wmr[k], av[j]);
                ag[j] = fmaf(mnc, wgr[k], ag[j]);
            }
        }

        #pragma unroll
        for (int j = 0; j < 8; ++j) {
            // k = lane + 32*j  ->  h = j, d = lane
            float gv = 1.0f / (1.0f + __expf(-ag[j]));
            size_t vidx = (((size_t)j * S_DIM + s) * D_DIM + lane) * N_DIM + n;
            vT[vidx] = (_Float16)av[j];
            size_t gidx = ((size_t)s * N_DIM + n) * HD_DIM + (lane + 32 * j);
            g16[gidx] = (_Float16)gv;
        }
    }
}

// ---------------------------------------------------------------------------
// Kernel 2: LayerNorm(z[p,:,:]) -> b[p,q,h] -> softmax over q -> wT[h][q][p]
// One block per p.
// ---------------------------------------------------------------------------
__global__ __launch_bounds__(256) void ln_softmax_z_kernel(
    const float* __restrict__ z, const float* __restrict__ nw,
    const float* __restrict__ nb, const float* __restrict__ wz,
    _Float16* __restrict__ wT)
{
    __shared__ float s_b[N_DIM * H_DIM];   // 24 KB
    __shared__ float s_wz[CZ_DIM * H_DIM]; // 4 KB
    __shared__ float s_nw[CZ_DIM], s_nb[CZ_DIM];
    __shared__ float s_red[256];

    const int t = threadIdx.x;
    const int p = blockIdx.x;

    for (int i = t; i < CZ_DIM * H_DIM; i += 256) s_wz[i] = wz[i];
    if (t < CZ_DIM) { s_nw[t] = nw[t]; s_nb[t] = nb[t]; }
    __syncthreads();

    for (int q = t; q < N_DIM; q += 256) {
        const float* row = z + ((size_t)p * N_DIM + q) * CZ_DIM;
        float sum = 0.f, ss = 0.f;
        for (int c = 0; c < CZ_DIM; ++c) { float x = row[c]; sum += x; ss += x * x; }
        float mu   = sum * (1.0f / CZ_DIM);
        float var  = ss * (1.0f / CZ_DIM) - mu * mu;
        float rstd = rsqrtf(var + EPSV);

        float acc[H_DIM];
        #pragma unroll
        for (int h = 0; h < H_DIM; ++h) acc[h] = 0.f;
        for (int c = 0; c < CZ_DIM; ++c) {
            float xn = (row[c] - mu) * rstd * s_nw[c] + s_nb[c];
            #pragma unroll
            for (int h = 0; h < H_DIM; ++h)
                acc[h] = fmaf(xn, s_wz[c * H_DIM + h], acc[h]);
        }
        #pragma unroll
        for (int h = 0; h < H_DIM; ++h) s_b[q * H_DIM + h] = acc[h];
    }
    __syncthreads();

    for (int h = 0; h < H_DIM; ++h) {
        float mx = -1e30f;
        for (int q = t; q < N_DIM; q += 256) mx = fmaxf(mx, s_b[q * H_DIM + h]);
        s_red[t] = mx; __syncthreads();
        for (int off = 128; off >= 1; off >>= 1) {
            if (t < off) s_red[t] = fmaxf(s_red[t], s_red[t + off]);
            __syncthreads();
        }
        mx = s_red[0]; __syncthreads();

        float e[3]; float lsum = 0.f; int qi = 0;
        for (int q = t; q < N_DIM; q += 256) {
            float ev = __expf(s_b[q * H_DIM + h] - mx);
            e[qi++] = ev; lsum += ev;
        }
        s_red[t] = lsum; __syncthreads();
        for (int off = 128; off >= 1; off >>= 1) {
            if (t < off) s_red[t] += s_red[t + off];
            __syncthreads();
        }
        float inv = 1.0f / s_red[0]; __syncthreads();

        qi = 0;
        for (int q = t; q < N_DIM; q += 256)
            wT[((size_t)h * N_DIM + q) * N_DIM + p] = (_Float16)(e[qi++] * inv);
    }
}

// ---------------------------------------------------------------------------
// Kernel 3: per-h GEMM  o[(s,d), q] = sum_p vT[h][(s,d)][p] * wT[h][q][p]
// M = S*D = 16384, N = 768, K = 768.
// Block tile 128x128, wave tile 64x32 (8 WMMA / K-step / wave), K-step 32.
// Async double-buffered global->LDS staging (ASYNCcnt).  Fused gate epilogue.
// ---------------------------------------------------------------------------
__global__ __launch_bounds__(256) void pav_gemm_kernel(
    const _Float16* __restrict__ vT, const _Float16* __restrict__ wT,
    const _Float16* __restrict__ g16, _Float16* __restrict__ o16)
{
    __shared__ _Float16 sA[2][128 * 32]; // 2 x 8 KB
    __shared__ _Float16 sB[2][128 * 32]; // 2 x 8 KB

    const int tid  = threadIdx.x;
    const int lane = tid & 31;
    const int wave = tid >> 5;
    const int wm_  = wave >> 2; // 0..1 : 64-row slab
    const int wn_  = wave & 3;  // 0..3 : 32-col slab
    const int h    = blockIdx.z;
    const int m0   = blockIdx.y * 128;
    const int n0   = blockIdx.x * 128;

    const _Float16* A = vT + (size_t)h * (S_DIM * D_DIM) * N_DIM;
    const _Float16* B = wT + (size_t)h * N_DIM * N_DIM;

    // per-thread staging chunks: 2 x (8 halves) for A, same for B
    const int lin0 = tid * 8;
    const int lin1 = (tid + 256) * 8;
    const int ar0 = lin0 >> 5, ac0 = lin0 & 31;
    const int ar1 = lin1 >> 5, ac1 = lin1 & 31;

    auto issue_tile = [&](int buf, int k0) {
        async_copy_b128(&A[(size_t)(m0 + ar0) * N_DIM + k0 + ac0], &sA[buf][lin0]);
        async_copy_b128(&A[(size_t)(m0 + ar1) * N_DIM + k0 + ac1], &sA[buf][lin1]);
        async_copy_b128(&B[(size_t)(n0 + ar0) * N_DIM + k0 + ac0], &sB[buf][lin0]);
        async_copy_b128(&B[(size_t)(n0 + ar1) * N_DIM + k0 + ac1], &sB[buf][lin1]);
    };

    v8f zero = {0.f, 0.f, 0.f, 0.f, 0.f, 0.f, 0.f, 0.f};
    v8f acc[4][2];
    #pragma unroll
    for (int i = 0; i < 4; ++i)
        #pragma unroll
        for (int j = 0; j < 2; ++j) acc[i][j] = zero;

    issue_tile(0, 0);

    const int KT = N_DIM / 32; // 24
    for (int kt = 0; kt < KT; ++kt) {
        const int cur = kt & 1;
        if (kt + 1 < KT) {
            issue_tile(cur ^ 1, (kt + 1) * 32);
            WAIT_ASYNC(4);  // 4 newest (next tile) may remain outstanding
        } else {
            WAIT_ASYNC(0);
        }
        __syncthreads();

        // A fragment: lane<16 holds K {0..15}, lane>=16 holds K {16..31}
        const int arow  = lane & 15;
        const int akoff = (lane >> 4) * 16;
        const int bkoff = (lane >> 4) * 16;
        v16h afrag[4], bfrag[2];
        #pragma unroll
        for (int i = 0; i < 4; ++i) {
            int mrow = wm_ * 64 + i * 16 + arow;
            ((v8h*)&afrag[i])[0] = *(const v8h*)&sA[cur][mrow * 32 + akoff];
            ((v8h*)&afrag[i])[1] = *(const v8h*)&sA[cur][mrow * 32 + akoff + 8];
        }
        #pragma unroll
        for (int j = 0; j < 2; ++j) {
            int col = wn_ * 32 + j * 16 + (lane & 15);
            bfrag[j] = *(const v16h*)&sB[cur][col * 32 + bkoff];
        }

        #pragma unroll
        for (int i = 0; i < 4; ++i)
            #pragma unroll
            for (int j = 0; j < 2; ++j)
                acc[i][j] = __builtin_amdgcn_wmma_f32_16x16x32_f16(
                    false, afrag[i], false, bfrag[j], (short)0, acc[i][j],
                    false, false);
        __syncthreads();
    }

    // Epilogue: C layout -> VGPR r: M = r + 8*(lane>>4), Ncol = lane&15
    const int mlocal = (lane >> 4) * 8;
    const int ncol   = lane & 15;
    #pragma unroll
    for (int i = 0; i < 4; ++i)
        #pragma unroll
        for (int j = 0; j < 2; ++j)
            #pragma unroll
            for (int r = 0; r < 8; ++r) {
                int mrow = m0 + wm_ * 64 + i * 16 + mlocal + r;
                int q    = n0 + wn_ * 32 + j * 16 + ncol;
                int s = mrow >> 5, d = mrow & 31;
                size_t idx = ((size_t)s * N_DIM + q) * HD_DIM + h * D_DIM + d;
                float gate = (float)g16[idx];
                o16[idx] = (_Float16)(acc[i][j][r] * gate);
            }
}

// ---------------------------------------------------------------------------
// Kernel 4: out[(s,q), c] = sum_k o16[(s,q)][k] * wo[k][c]
// M = S*N = 393216, N = 64, K = 256.  wo staged to LDS as f16 once per block;
// A tiles double-buffered via async global->LDS.
// ---------------------------------------------------------------------------
__global__ __launch_bounds__(256) void out_gemm_kernel(
    const _Float16* __restrict__ o16, const float* __restrict__ wo,
    float* __restrict__ out)
{
    __shared__ _Float16 sA[2][128 * 32];   // 2 x 8 KB
    __shared__ _Float16 sW[64 * 256];      // 32 KB, [col][k]

    const int tid  = threadIdx.x;
    const int lane = tid & 31;
    const int wave = tid >> 5;
    const int wm_  = wave >> 1; // 0..3
    const int wn_  = wave & 1;  // 0..1
    const int r0   = blockIdx.x * 128;

    const int lin0 = tid * 8;
    const int lin1 = (tid + 256) * 8;
    const int ar0 = lin0 >> 5, ac0 = lin0 & 31;
    const int ar1 = lin1 >> 5, ac1 = lin1 & 31;

    auto issue_tile = [&](int buf, int k0) {
        async_copy_b128(&o16[(size_t)(r0 + ar0) * HD_DIM + k0 + ac0], &sA[buf][lin0]);
        async_copy_b128(&o16[(size_t)(r0 + ar1) * HD_DIM + k0 + ac1], &sA[buf][lin1]);
    };

    issue_tile(0, 0);

    for (int i = tid; i < CM_DIM * HD_DIM; i += 256) {
        int col = i >> 8;   // / 256
        int k   = i & 255;
        sW[i] = (_Float16)wo[(size_t)k * CM_DIM + col];
    }

    v8f zero = {0.f, 0.f, 0.f, 0.f, 0.f, 0.f, 0.f, 0.f};
    v8f acc[2][2];
    #pragma unroll
    for (int i = 0; i < 2; ++i)
        #pragma unroll
        for (int j = 0; j < 2; ++j) acc[i][j] = zero;

    const int KT = HD_DIM / 32; // 8
    for (int kt = 0; kt < KT; ++kt) {
        const int cur = kt & 1;
        if (kt + 1 < KT) {
            issue_tile(cur ^ 1, (kt + 1) * 32);
            WAIT_ASYNC(2);
        } else {
            WAIT_ASYNC(0);
        }
        __syncthreads();

        const int arow  = lane & 15;
        const int akoff = (lane >> 4) * 16;
        const int bkoff = (lane >> 4) * 16;
        v16h afrag[2], bfrag[2];
        #pragma unroll
        for (int i = 0; i < 2; ++i) {
            int mrow = wm_ * 32 + i * 16 + arow;
            ((v8h*)&afrag[i])[0] = *(const v8h*)&sA[cur][mrow * 32 + akoff];
            ((v8h*)&afrag[i])[1] = *(const v8h*)&sA[cur][mrow * 32 + akoff + 8];
        }
        #pragma unroll
        for (int j = 0; j < 2; ++j) {
            int col = wn_ * 32 + j * 16 + (lane & 15);
            bfrag[j] = *(const v16h*)&sW[col * 256 + kt * 32 + bkoff];
        }

        #pragma unroll
        for (int i = 0; i < 2; ++i)
            #pragma unroll
            for (int j = 0; j < 2; ++j)
                acc[i][j] = __builtin_amdgcn_wmma_f32_16x16x32_f16(
                    false, afrag[i], false, bfrag[j], (short)0, acc[i][j],
                    false, false);
        __syncthreads();
    }

    const int mlocal = (lane >> 4) * 8;
    const int ncol   = lane & 15;
    #pragma unroll
    for (int i = 0; i < 2; ++i)
        #pragma unroll
        for (int j = 0; j < 2; ++j)
            #pragma unroll
            for (int r = 0; r < 8; ++r) {
                int row = r0 + wm_ * 32 + i * 16 + mlocal + r;
                int col = wn_ * 32 + j * 16 + ncol;
                out[(size_t)row * CM_DIM + col] = acc[i][j][r];
            }
}

// ---------------------------------------------------------------------------
extern "C" void kernel_launch(void* const* d_in, const int* in_sizes, int n_in,
                              void* d_out, int out_size, void* d_ws, size_t ws_size,
                              hipStream_t stream)
{
    (void)in_sizes; (void)n_in; (void)out_size; (void)ws_size;

    const float* m   = (const float*)d_in[0];
    const float* z   = (const float*)d_in[1];
    const float* nmw = (const float*)d_in[2];
    const float* nmb = (const float*)d_in[3];
    const float* nzw = (const float*)d_in[4];
    const float* nzb = (const float*)d_in[5];
    const float* wm  = (const float*)d_in[6];
    const float* wg  = (const float*)d_in[7];
    const float* wz  = (const float*)d_in[8];
    const float* wo  = (const float*)d_in[9];
    float* out = (float*)d_out;

    char* ws = (char*)d_ws;
    size_t off = 0;
    _Float16* vT  = (_Float16*)(ws + off);
    off += (size_t)H_DIM * S_DIM * D_DIM * N_DIM * sizeof(_Float16);
    _Float16* g16 = (_Float16*)(ws + off);
    off += (size_t)S_DIM * N_DIM * HD_DIM * sizeof(_Float16);
    _Float16* wT  = (_Float16*)(ws + off);
    off += (size_t)H_DIM * N_DIM * N_DIM * sizeof(_Float16);
    _Float16* o16 = (_Float16*)(ws + off);
    off += (size_t)S_DIM * N_DIM * HD_DIM * sizeof(_Float16);

    ln_project_m_kernel<<<dim3(S_DIM), dim3(256), 0, stream>>>(
        m, nmw, nmb, wm, wg, vT, g16);

    ln_softmax_z_kernel<<<dim3(N_DIM), dim3(256), 0, stream>>>(
        z, nzw, nzb, wz, wT);

    pav_gemm_kernel<<<dim3(N_DIM / 128, (S_DIM * D_DIM) / 128, H_DIM),
                      dim3(256), 0, stream>>>(vT, wT, g16, o16);

    out_gemm_kernel<<<dim3((S_DIM * N_DIM) / 128), dim3(256), 0, stream>>>(
        o16, wo, out);
}
